// HierarchicalCompositionalModel_87505663688808
// MI455X (gfx1250) — matrix operand order repaired and checked by reading, not compiled, not run on hardware
//
#include <hip/hip_runtime.h>

typedef __attribute__((ext_vector_type(16))) __bf16 v16bf;
typedef __attribute__((ext_vector_type(8)))  float  v8f;
typedef __attribute__((ext_vector_type(4)))  float  f32x4;
typedef __attribute__((ext_vector_type(4)))  __bf16 bf16x4;
typedef __attribute__((ext_vector_type(4)))  int    v4i;
typedef __attribute__((ext_vector_type(8)))  int    v8i;
typedef __attribute__((ext_vector_type(4)))  unsigned v4u;

#define B_DIM 4096
#define D_DIM 4096
#define H_DIM 2048
#define T_STEPS 8
#define M_MOD 6

#define TM 128
#define TN 128
#define TK 32
#define LDSTR 40   // bf16 elems per LDS row: 20 dwords -> 16 row offsets distinct mod 64 banks

// ---- CDNA5 data-mover feature detection ----
#if __has_builtin(__builtin_amdgcn_tensor_load_to_lds)
#define HAVE_TDM 1
#else
#define HAVE_TDM 0
#endif
#if __has_builtin(__builtin_amdgcn_global_load_async_to_lds_b128)
#define HAVE_ASYNC_LD 1
#else
#define HAVE_ASYNC_LD 0
#endif
#if __has_builtin(__builtin_amdgcn_global_store_async_from_lds_b128)
#define HAVE_ASYNC_ST 1
#else
#define HAVE_ASYNC_ST 0
#endif

#define AS_GLOBAL __attribute__((address_space(1)))
#define AS_LOCAL  __attribute__((address_space(3)))
// builtin wants non-const v4i pointers: global (AS1 / __device__) and LDS (AS3 / __shared__)
#define GASYNC(p) ((AS_GLOBAL v4i*)(uintptr_t)(p))
#define LASYNC(p) ((AS_LOCAL v4i*)(unsigned)(uintptr_t)(p))

__device__ __forceinline__ void wait_asynccnt0() {
#if __has_builtin(__builtin_amdgcn_s_wait_asynccnt)
  __builtin_amdgcn_s_wait_asynccnt(0);
#else
  asm volatile("s_wait_asynccnt 0x0" ::: "memory");
#endif
}
__device__ __forceinline__ void wait_tensorcnt0() {
#if __has_builtin(__builtin_amdgcn_s_wait_tensorcnt)
  __builtin_amdgcn_s_wait_tensorcnt(0);
#else
  asm volatile("s_wait_tensorcnt 0x0" ::: "memory");
#endif
}

#if HAVE_TDM
// One TDM op: 128-row x 32-elem bf16 tile, global row stride kd elems, into LDS
// at lds_off with 4-dword pad after every 16 data dwords => LDS row stride 40 bf16.
// D# fields per CDNA5 ISA 8.3-8.5. This toolchain exposes the 6-arg builtin:
// (uint32x4 g0, int32x8 g1, int32x4 g2, int32x4 g3, int32x8 g4, i32 cpol).
__device__ __forceinline__ void tdm_load_tile_bf16(const __bf16* gsrc, unsigned lds_off, int kd)
{
  unsigned long long ga = (unsigned long long)(uintptr_t)gsrc;
  v4u g0 = { 1u,                                           // count=1, user mode
             lds_off,                                      // lds_addr (bytes)
             (unsigned)(ga & 0xFFFFFFFFu),                 // global_addr[31:0]
             (unsigned)((ga >> 32) & 0x01FFFFFFu) | (2u << 30) };  // addr[56:32] | type=2
  v8i g1 = { (int)((1u << 16) | (1u << 20) | (3u << 22) | (3u << 25)),
             // wg_mask=0 (not in cluster), data_size=1 (2B), pad_enable,
             // pad_interval=3 (16 dwords), pad_amount=3 (4 dwords)
             (int)(((unsigned)kd & 0xFFFFu) << 16),        // abar=0 | tensor_dim0[15:0]
             (int)((((unsigned)kd >> 16) & 0xFFFFu) | (((unsigned)B_DIM & 0xFFFFu) << 16)),
             // tensor_dim0[31:16] | tensor_dim1[15:0]
             (int)((((unsigned)B_DIM >> 16) & 0xFFFFu) | (32u << 16)),
             // tensor_dim1[31:16] | tile_dim0=32
             (int)(128u),                                  // tile_dim1=128 | tile_dim2=0
             (int)kd,                                      // tensor_dim0_stride[31:0]
             0,                                            // stride0 hi | dim1_stride lo
             0 };                                          // dim1_stride hi
  v4i g2 = { 0, 0, 0, 0 };
  v4i g3 = { 0, 0, 0, 0 };
  v8i g4 = { 0, 0, 0, 0, 0, 0, 0, 0 };
  __builtin_amdgcn_tensor_load_to_lds(g0, g1, g2, g3, g4, 0);
}
#endif

// ---------------- f32 -> bf16 (vectorized x4) ----------------
__global__ __launch_bounds__(256) void cvt_bf16_kernel(const float* __restrict__ in,
                                                       __bf16* __restrict__ out, int n4) {
  int i = blockIdx.x * 256 + threadIdx.x;
  if (i < n4) {
    f32x4 v = ((const f32x4*)in)[i];
    bf16x4 o;
    o[0] = (__bf16)v[0]; o[1] = (__bf16)v[1]; o[2] = (__bf16)v[2]; o[3] = (__bf16)v[3];
    ((bf16x4*)out)[i] = o;
  }
}

// ---------------- bf16 WMMA GEMM: C = [relu](A @ B + bias) ----------------
// A: [M,KD] bf16 row-major; B: [KD,ND] bf16 row-major; C: [M,ND] bf16.
// 128x128 block tile, K-step 32, double-buffered LDS, TDM/async A staging.
template<int KD, int ND>
__global__ __launch_bounds__(256) void gemm_bf16_wmma_kernel(
    const __bf16* __restrict__ A, const __bf16* __restrict__ Bm,
    const float* __restrict__ bias, __bf16* __restrict__ C, int do_relu)
{
  __shared__ __bf16 As[2][TM * LDSTR];   // [row][k]   2 x 10 KB
  __shared__ __bf16 Bs[2][TN * LDSTR];   // [n][k]     2 x 10 KB (transposed)

  const int tid   = threadIdx.x;
  const int lane  = tid & 31;
  const int wave  = tid >> 5;      // 8 waves
  const int waveM = wave & 3;      // 4 slabs of 32 rows
  const int waveN = wave >> 2;     // 2 slabs of 64 cols
  const int l15   = lane & 15;
  const int hi    = lane >> 4;

  const int bm = blockIdx.y * TM;
  const int bn = blockIdx.x * TN;

  v8f acc[2][4] = {};

  auto stageA = [&](int buf, int k0) {
#if HAVE_TDM
    if (wave == 0)   // TDM ignores EXEC; one descriptor moves the whole 128x32 tile
      tdm_load_tile_bf16(A + (size_t)bm * KD + k0, (unsigned)(uintptr_t)&As[buf][0], KD);
#else
    #pragma unroll
    for (int u = 0; u < 2; ++u) {                   // 128 rows x 4 x 16B chunks
      int c    = tid + u * 256;
      int row  = c >> 2;
      int col8 = (c & 3) << 3;
      const __bf16* g = A + (size_t)(bm + row) * KD + k0 + col8;
      __bf16* l = &As[buf][row * LDSTR + col8];
#if HAVE_ASYNC_LD
      __builtin_amdgcn_global_load_async_to_lds_b128(GASYNC(g), LASYNC(l), 0, 0);
#else
      *(f32x4*)l = *(const f32x4*)g;
#endif
    }
#endif
  };

  auto stageB = [&](int buf, int k0) {              // 32 x 128 -> Bs[n][k]
    #pragma unroll
    for (int u = 0; u < 2; ++u) {
      int c    = tid + u * 256;
      int kk   = c >> 4;
      int col8 = (c & 15) << 3;
      f32x4 d = *(const f32x4*)(Bm + (size_t)(k0 + kk) * ND + bn + col8);
      __bf16 tmp[8] __attribute__((aligned(16)));
      *(f32x4*)tmp = d;
      #pragma unroll
      for (int i = 0; i < 8; ++i) Bs[buf][(col8 + i) * LDSTR + kk] = tmp[i];
    }
  };

  auto compute = [&](int buf) {
    // A fragment: lane half hi holds K = hi*8..+7 (e0..7), 16+hi*8..+7 (e8..15)
    v16bf afrag[2], bfrag[4];
    #pragma unroll
    for (int mt = 0; mt < 2; ++mt) {
      const __bf16* p = &As[buf][(waveM * 32 + mt * 16 + l15) * LDSTR + hi * 8];
      ((f32x4*)&afrag[mt])[0] = *(const f32x4*)p;
      ((f32x4*)&afrag[mt])[1] = *(const f32x4*)(p + 16);
    }
    // B fragment: lane half hi holds K = hi*8..+7 (e0..7), 16+hi*8..+7 (e8..15)
    #pragma unroll
    for (int nt = 0; nt < 4; ++nt) {
      const __bf16* p = &Bs[buf][(waveN * 64 + nt * 16 + l15) * LDSTR + hi * 8];
      ((f32x4*)&bfrag[nt])[0] = *(const f32x4*)p;
      ((f32x4*)&bfrag[nt])[1] = *(const f32x4*)(p + 16);
    }
    #pragma unroll
    for (int mt = 0; mt < 2; ++mt)
      #pragma unroll
      for (int nt = 0; nt < 4; ++nt)
        acc[mt][nt] = __builtin_amdgcn_wmma_f32_16x16x32_bf16(
            false, afrag[mt], false, bfrag[nt], (short)0, acc[mt][nt], false, false);
  };

  constexpr int nK = KD / TK;
  stageA(0, 0);
  stageB(0, 0);
#if HAVE_TDM
  wait_tensorcnt0();
#elif HAVE_ASYNC_LD
  wait_asynccnt0();
#endif
  __syncthreads();

  for (int kt = 0; kt < nK; ++kt) {
    const int cur = kt & 1;
    if (kt + 1 < nK) {                 // prefetch next tile into the other buffer
      stageA(cur ^ 1, (kt + 1) * TK);
      stageB(cur ^ 1, (kt + 1) * TK);
    }
    compute(cur);
#if HAVE_TDM
    wait_tensorcnt0();                 // no-op for non-issuing waves (TENSORcnt==0)
#elif HAVE_ASYNC_LD
    wait_asynccnt0();
#endif
    __syncthreads();
  }

  // Epilogue: C/D layout — VGPR r: M = r + 8*hi, N = lane&15
  #pragma unroll
  for (int mt = 0; mt < 2; ++mt) {
    #pragma unroll
    for (int nt = 0; nt < 4; ++nt) {
      int col  = bn + waveN * 64 + nt * 16 + l15;
      int row0 = bm + waveM * 32 + mt * 16 + hi * 8;
      float bv = bias[col];
      #pragma unroll
      for (int r = 0; r < 8; ++r) {
        float v = acc[mt][nt][r] + bv;
        if (do_relu) v = fmaxf(v, 0.0f);
        C[(size_t)(row0 + r) * ND + col] = (__bf16)v;
      }
    }
  }
}

// ---------------- logits (enc @ Ws + bs) + per-step argmax routing ----------------
__global__ __launch_bounds__(64) void logits_argmax_kernel(
    const __bf16* __restrict__ enc, const float* __restrict__ Ws,
    const float* __restrict__ bs, float* __restrict__ logits_out,
    int* __restrict__ idx_out)
{
  const int b = blockIdx.x;
  const int t = threadIdx.x;
  __shared__ float lg[T_STEPS * M_MOD];

  if (t < T_STEPS * M_MOD) {
    float a = bs[t];
    const __bf16* er = enc + (size_t)b * H_DIM;
    for (int k = 0; k < H_DIM; ++k)
      a += (float)er[k] * Ws[(size_t)k * (T_STEPS * M_MOD) + t];
    lg[t] = a;
    logits_out[(size_t)b * (T_STEPS * M_MOD) + t] = a;
  }
  __syncthreads();
  if (t < T_STEPS) {
    float best = lg[t * M_MOD];
    int bi = 0;
    for (int m = 1; m < M_MOD; ++m) {
      float v = lg[t * M_MOD + m];
      if (v > best) { best = v; bi = m; }   // first-max tie-break like jnp.argmax
    }
    idx_out[b * T_STEPS + t] = bi;
  }
}

// ---------------- sequential module scan (sort/reverse/add/sub/mul/div) ----------------
__global__ __launch_bounds__(512) void scan_kernel(
    const float* __restrict__ x, const int* __restrict__ idx,
    const float* __restrict__ add_p, const float* __restrict__ sub_p,
    const float* __restrict__ mul_p, const float* __restrict__ div_p,
    float* __restrict__ out)
{
  __shared__ float cur[D_DIM];   // 16 KB
  const int b   = blockIdx.x;
  const int tid = threadIdx.x;

  const float addv = add_p[0];
  const float subv = sub_p[0];
  const float mulv = mul_p[0];
  const float divv = div_p[0] + 1e-5f;

  const float* xr = x + (size_t)b * D_DIM;
  for (int i = tid; i < D_DIM; i += 512) cur[i] = xr[i];
  __syncthreads();

  for (int t = 0; t < T_STEPS; ++t) {
    const int m = idx[b * T_STEPS + t];   // uniform across the block
    if (m == 0) {
      // bitonic sort ascending, in LDS
      for (int k = 2; k <= D_DIM; k <<= 1) {
        for (int j = k >> 1; j > 0; j >>= 1) {
          for (int i = tid; i < D_DIM; i += 512) {
            int ixj = i ^ j;
            if (ixj > i) {
              float a = cur[i], c = cur[ixj];
              bool up = ((i & k) == 0);
              if ((a > c) == up) { cur[i] = c; cur[ixj] = a; }
            }
          }
          __syncthreads();
        }
      }
    } else if (m == 1) {
      for (int i = tid; i < D_DIM / 2; i += 512) {
        float a = cur[i], c = cur[D_DIM - 1 - i];
        cur[i] = c; cur[D_DIM - 1 - i] = a;
      }
      __syncthreads();
    } else {
      for (int i = tid; i < D_DIM; i += 512) {
        float v = cur[i];
        if      (m == 2) v += addv;
        else if (m == 3) v -= subv;
        else if (m == 4) v *= mulv;
        else             v = v / divv;
        cur[i] = v;
      }
      __syncthreads();
    }
    float* orow = out + ((size_t)b * T_STEPS + t) * D_DIM;
#if HAVE_ASYNC_ST
    // stream 16 KB row LDS -> HBM on the async path (no VGPR round trip)
    #pragma unroll
    for (int u = 0; u < 2; ++u) {
      int c4 = (tid + u * 512) * 4;
      __builtin_amdgcn_global_store_async_from_lds_b128(GASYNC(orow + c4), LASYNC(&cur[c4]), 0, 0);
    }
    wait_asynccnt0();      // LDS reads done before next step mutates cur
    __syncthreads();
#else
    for (int i = tid; i < D_DIM; i += 512) orow[i] = cur[i];
    __syncthreads();
#endif
  }
}

extern "C" void kernel_launch(void* const* d_in, const int* in_sizes, int n_in,
                              void* d_out, int out_size, void* d_ws, size_t ws_size,
                              hipStream_t stream) {
  (void)in_sizes; (void)n_in; (void)out_size; (void)ws_size;
  const float* x   = (const float*)d_in[0];
  const float* W1  = (const float*)d_in[1];
  const float* b1  = (const float*)d_in[2];
  const float* W2  = (const float*)d_in[3];
  const float* b2  = (const float*)d_in[4];
  const float* Ws  = (const float*)d_in[5];
  const float* bs  = (const float*)d_in[6];
  const float* add_p = (const float*)d_in[7];
  const float* sub_p = (const float*)d_in[8];
  const float* mul_p = (const float*)d_in[9];
  const float* div_p = (const float*)d_in[10];

  // Workspace layout (bf16 stages + routing indices), ~92.4 MB
  __bf16* xb  = (__bf16*)d_ws;
  __bf16* w1b = xb  + (size_t)B_DIM * D_DIM;
  __bf16* w2b = w1b + (size_t)D_DIM * H_DIM;
  __bf16* hb  = w2b + (size_t)H_DIM * H_DIM;
  __bf16* eb  = hb  + (size_t)B_DIM * H_DIM;
  int*    idx = (int*)(eb + (size_t)B_DIM * H_DIM);

  float* out    = (float*)d_out;
  float* logits = out + (size_t)B_DIM * T_STEPS * D_DIM;

  // 1) convert inputs to bf16
  {
    int n4;
    n4 = B_DIM * D_DIM / 4; cvt_bf16_kernel<<<(n4 + 255) / 256, 256, 0, stream>>>(x,  xb,  n4);
    n4 = D_DIM * H_DIM / 4; cvt_bf16_kernel<<<(n4 + 255) / 256, 256, 0, stream>>>(W1, w1b, n4);
    n4 = H_DIM * H_DIM / 4; cvt_bf16_kernel<<<(n4 + 255) / 256, 256, 0, stream>>>(W2, w2b, n4);
  }

  // 2) h = relu(x @ W1 + b1)   [4096 x 2048], K = 4096
  gemm_bf16_wmma_kernel<D_DIM, H_DIM><<<dim3(H_DIM / TN, B_DIM / TM), 256, 0, stream>>>(
      xb, w1b, b1, hb, 1);

  // 3) enc = h @ W2 + b2       [4096 x 2048], K = 2048
  gemm_bf16_wmma_kernel<H_DIM, H_DIM><<<dim3(H_DIM / TN, B_DIM / TM), 256, 0, stream>>>(
      hb, w2b, b2, eb, 0);

  // 4) logits = enc @ Ws + bs; argmax over M per (b,t)
  logits_argmax_kernel<<<B_DIM, 64, 0, stream>>>(eb, Ws, bs, logits, idx);

  // 5) routed module scan -> outputs [B, T, D]
  scan_kernel<<<B_DIM, 512, 0, stream>>>(x, idx, add_p, sub_p, mul_p, div_p, out);
}
